// BiLstmCellEncoder_9792525435320
// MI455X (gfx1250) — matrix operand order repaired and checked by reading, not compiled, not run on hardware
//
#include <hip/hip_runtime.h>
#include <hip/hip_bf16.h>

// ---------------------------------------------------------------------------
// Problem constants (B,S,D,H,A) = (128, 512, 256, 256, 64)
// ---------------------------------------------------------------------------
#define BB   128
#define SS   512
#define DD   256
#define HH   256
#define G4H  1024   // 4*H
#define AA   64

typedef __attribute__((ext_vector_type(16))) __bf16          v16bf;
typedef __attribute__((ext_vector_type(8)))  float           v8f;
typedef __attribute__((ext_vector_type(8)))  unsigned short  ushort8;

// ---------------- bf16 <-> f32 helpers (bit-exact RNE, no libcalls) --------
__device__ __forceinline__ float bf2f(unsigned short u) {
  unsigned v = ((unsigned)u) << 16;
  return __builtin_bit_cast(float, v);
}
__device__ __forceinline__ unsigned short f2bfbits(float f) {
  unsigned u = __builtin_bit_cast(unsigned, f);
  unsigned r = u + 0x7FFFu + ((u >> 16) & 1u);   // round-to-nearest-even
  return (unsigned short)(r >> 16);
}
__device__ __forceinline__ float sigmoidf(float x) {
  return 1.0f / (1.0f + __expf(-x));
}

// ---------------------------------------------------------------------------
// Core: accumulate C += Arow(1x256 bf16) x Wrow-major(256) via 8x WMMA K=32.
//   A layout (16x32 bf16, wave32): lane l (m = l&15, half = l>>4):
//     element j: k = 16*half + j
//   B layout (32x16 bf16): lane l (n = l&15): element j: k = 16*half + j
//   Row-major weight W[n][k] == (W^T)[k][n], so B's 16 contiguous k-elements
//   are 16 contiguous bf16 of weight row n -> two 16B global loads per lane.
// ---------------------------------------------------------------------------
__device__ __forceinline__ v8f wmma_row_k256(const unsigned short* __restrict__ arow,
                                             const unsigned short* __restrict__ wrow,
                                             int half, v8f c) {
#pragma unroll
  for (int kb = 0; kb < 256; kb += 32) {
    ushort8 a0 = *(const ushort8*)(arow + kb + 16 * half);
    ushort8 a1 = *(const ushort8*)(arow + kb + 16 * half + 8);
    ushort8 b0 = *(const ushort8*)(wrow + kb + 16 * half);
    ushort8 b1 = *(const ushort8*)(wrow + kb + 16 * half + 8);
    v16bf A, Bv;
#pragma unroll
    for (int j = 0; j < 8; ++j) {
      A[j]      = __builtin_bit_cast(__bf16, (unsigned short)a0[j]);
      A[j + 8]  = __builtin_bit_cast(__bf16, (unsigned short)a1[j]);
      Bv[j]     = __builtin_bit_cast(__bf16, (unsigned short)b0[j]);
      Bv[j + 8] = __builtin_bit_cast(__bf16, (unsigned short)b1[j]);
    }
    c = __builtin_amdgcn_wmma_f32_16x16x32_bf16(false, A, false, Bv,
                                                (short)0, c, false, false);
  }
  return c;
}

// ---------------------------------------------------------------------------
// One recurrent step, both directions fused:
//   G[dir][b][0:4H] = x_t @ Wih^T + h_prev @ Whh^T + (bih+bhh)
// 1024 waves: dir(2) x Mtile(8) x Ntile(64); one 16x16 f32 tile per wave.
// ---------------------------------------------------------------------------
__global__ void __launch_bounds__(256)
k_lstm_gemm(const unsigned short* __restrict__ Xf, const unsigned short* __restrict__ Xb,
            int tf, int tb,
            const unsigned short* __restrict__ Wih_f, const unsigned short* __restrict__ Whh_f,
            const unsigned short* __restrict__ Wih_b, const unsigned short* __restrict__ Whh_b,
            const float* __restrict__ bias_f, const float* __restrict__ bias_b,
            const unsigned short* __restrict__ Hprev,   // [2][B][H] bf16
            float* __restrict__ G)                      // [2][B][4H] f32
{
  int wid  = blockIdx.x * 8 + (threadIdx.x >> 5);   // 0..1023
  int lane = threadIdx.x & 31;
  int dir  = wid >> 9;            // 0 = fwd, 1 = bwd
  int tile = wid & 511;
  int mt   = tile >> 6;           // batch tile   0..7
  int nt   = tile & 63;           // gate-col tile 0..63
  int m16  = lane & 15;
  int half = lane >> 4;

  const unsigned short* X   = dir ? Xb    : Xf;
  const unsigned short* Wih = dir ? Wih_b : Wih_f;
  const unsigned short* Whh = dir ? Whh_b : Whh_f;
  const float*          bia = dir ? bias_b : bias_f;
  int td = dir ? tb : tf;

  int mrow = mt * 16 + m16;       // batch row for A
  int ncol = nt * 16 + m16;       // output column / weight row for B

  const unsigned short* arow_x  = X     + ((size_t)mrow * SS + td) * HH;
  const unsigned short* arow_h  = Hprev + (size_t)(dir * BB + mrow) * HH;
  const unsigned short* wrow_ih = Wih   + (size_t)ncol * HH;
  const unsigned short* wrow_hh = Whh   + (size_t)ncol * HH;

  v8f c = {};
  c = wmma_row_k256(arow_x, wrow_ih, half, c);   // input contribution
  c = wmma_row_k256(arow_h, wrow_hh, half, c);   // recurrent contribution

  float bv = bia[ncol];
  float* gout = G + (size_t)(dir * BB) * G4H;
#pragma unroll
  for (int r = 0; r < 8; ++r) {                  // C layout: VGPR r -> M = r + 8*half
    int mr = mt * 16 + r + 8 * half;
    gout[(size_t)mr * G4H + ncol] = c[r] + bv;
  }
}

// ---------------------------------------------------------------------------
// Elementwise LSTM cell update (both directions): c,h update + bf16 emit.
// ---------------------------------------------------------------------------
__global__ void __launch_bounds__(256)
k_lstm_cell(const float* __restrict__ G, float* __restrict__ Cst,
            unsigned short* __restrict__ Hprev,
            unsigned short* __restrict__ hs_fout, unsigned short* __restrict__ hs_bout,
            int tf, int tb)
{
  int idx = blockIdx.x * 256 + threadIdx.x;   // 0 .. 2*B*H-1
  int dir = idx >> 15;                        // B*H = 32768
  int rem = idx & 32767;
  int b   = rem >> 8;
  int hj  = rem & 255;

  const float* g = G + (size_t)(dir * BB + b) * G4H;
  float gi = g[hj], gf = g[256 + hj], gg = g[512 + hj], go = g[768 + hj];

  float c  = Cst[idx];
  float cn = sigmoidf(gf) * c + sigmoidf(gi) * tanhf(gg);
  float h  = sigmoidf(go) * tanhf(cn);
  Cst[idx] = cn;

  unsigned short hb = f2bfbits(h);
  Hprev[idx] = hb;
  unsigned short* hsout = dir ? hs_bout : hs_fout;
  int td = dir ? tb : tf;
  hsout[((size_t)b * SS + td) * HH + hj] = hb;
}

// ---------------------------------------------------------------------------
// Attention scores: scores[b,s] = Wa2 . tanh(Wa1 @ out[b,s,:])
// ---------------------------------------------------------------------------
__global__ void __launch_bounds__(64)
k_attn_scores(const unsigned short* __restrict__ hsf, const unsigned short* __restrict__ hsb,
              const float* __restrict__ Wa1, const float* __restrict__ Wa2,
              float* __restrict__ scores)
{
  int bs = blockIdx.x;             // b*S + s
  int a  = threadIdx.x;            // 0..63
  const unsigned short* pf = hsf + (size_t)bs * HH;
  const unsigned short* pb = hsb + (size_t)bs * HH;
  const float* w = Wa1 + (size_t)a * (2 * HH);
  float acc = 0.f;
  for (int k = 0; k < HH; ++k) acc += bf2f(pf[k]) * w[k];
  for (int k = 0; k < HH; ++k) acc += bf2f(pb[k]) * w[HH + k];
  float v = tanhf(acc) * Wa2[a];
  __shared__ float red[64];
  red[a] = v; __syncthreads();
  for (int off = 32; off > 0; off >>= 1) {
    if (a < off) red[a] += red[a + off];
    __syncthreads();
  }
  if (a == 0) scores[bs] = red[0];
}

// ---------------------------------------------------------------------------
// Softmax over S + weighted pooling -> out[b, 0:2H]
// ---------------------------------------------------------------------------
__global__ void __launch_bounds__(512)
k_attn_pool(const float* __restrict__ scores,
            const unsigned short* __restrict__ hsf, const unsigned short* __restrict__ hsb,
            float* __restrict__ out)
{
  int b = blockIdx.x;
  int j = threadIdx.x;             // 0..511
  __shared__ float sw[512], red[512];
  float sc = scores[(size_t)b * SS + j];
  red[j] = sc; __syncthreads();
  for (int off = 256; off > 0; off >>= 1) {
    if (j < off) red[j] = fmaxf(red[j], red[j + off]);
    __syncthreads();
  }
  float m = red[0]; __syncthreads();
  float e = __expf(sc - m);
  red[j] = e; __syncthreads();
  for (int off = 256; off > 0; off >>= 1) {
    if (j < off) red[j] += red[j + off];
    __syncthreads();
  }
  float inv = 1.0f / red[0];
  sw[j] = e * inv; __syncthreads();

  const unsigned short* hs = (j < HH) ? hsf : hsb;
  int jj = j & 255;
  const unsigned short* base = hs + (size_t)b * SS * HH + jj;
  float acc = 0.f;
  for (int t = 0; t < SS; ++t) acc += sw[t] * bf2f(base[(size_t)t * HH]);
  out[(size_t)b * (2 * HH) + j] = acc;
}

// ---------------------------------------------------------------------------
// Utility kernels
// ---------------------------------------------------------------------------
__global__ void k_f32_to_bf16(const float* __restrict__ in,
                              unsigned short* __restrict__ outp, int n) {
  int i = blockIdx.x * 256 + threadIdx.x;
  if (i < n) outp[i] = f2bfbits(in[i]);
}
__global__ void k_add_vec(const float* __restrict__ a, const float* __restrict__ b,
                          float* __restrict__ o, int n) {
  int i = blockIdx.x * 256 + threadIdx.x;
  if (i < n) o[i] = a[i] + b[i];
}

// ---------------------------------------------------------------------------
// Host driver (graph-capture safe: kernel launches + hipMemsetAsync only)
// ---------------------------------------------------------------------------
extern "C" void kernel_launch(void* const* d_in, const int* in_sizes, int n_in,
                              void* d_out, int out_size, void* d_ws, size_t ws_size,
                              hipStream_t stream)
{
  (void)in_sizes; (void)n_in; (void)out_size; (void)ws_size;

  const float* x      = (const float*)d_in[0];
  const float* W_in[8] = {         // order: ih_f0, hh_f0, ih_b0, hh_b0, ih_f1, hh_f1, ih_b1, hh_b1
    (const float*)d_in[1],  (const float*)d_in[2],
    (const float*)d_in[9],  (const float*)d_in[10],
    (const float*)d_in[5],  (const float*)d_in[6],
    (const float*)d_in[13], (const float*)d_in[14]
  };
  const float* bih[4] = { (const float*)d_in[3],  (const float*)d_in[11],
                          (const float*)d_in[7],  (const float*)d_in[15] }; // f0,b0,f1,b1
  const float* bhh[4] = { (const float*)d_in[4],  (const float*)d_in[12],
                          (const float*)d_in[8],  (const float*)d_in[16] };
  const float* Wa1 = (const float*)d_in[17];
  const float* Wa2 = (const float*)d_in[18];

  // ---- carve workspace (256B aligned slices) ----
  char* p = (char*)d_ws;
  auto carve = [&](size_t bytes) -> void* {
    void* r = (void*)p;
    p += (bytes + 255) & ~(size_t)255;
    return r;
  };
  unsigned short* x_bf = (unsigned short*)carve((size_t)BB * SS * DD * 2);
  unsigned short* w_bf[8];
  for (int i = 0; i < 8; ++i) w_bf[i] = (unsigned short*)carve((size_t)G4H * HH * 2);
  float* bias_all = (float*)carve((size_t)4 * G4H * 4);          // f0,b0,f1,b1
  unsigned short* hs0f = (unsigned short*)carve((size_t)BB * SS * HH * 2);
  unsigned short* hs0b = (unsigned short*)carve((size_t)BB * SS * HH * 2);
  unsigned short* hs1f = (unsigned short*)carve((size_t)BB * SS * HH * 2);
  unsigned short* hs1b = (unsigned short*)carve((size_t)BB * SS * HH * 2);
  float*          G    = (float*)carve((size_t)2 * BB * G4H * 4);
  unsigned short* Hp   = (unsigned short*)carve((size_t)2 * BB * HH * 2);
  float*          Cst  = (float*)carve((size_t)2 * BB * HH * 4);
  float*          sc   = (float*)carve((size_t)BB * SS * 4);

  // ---- precision conversion (once per call) ----
  {
    int n = BB * SS * DD;
    k_f32_to_bf16<<<(n + 255) / 256, 256, 0, stream>>>(x, x_bf, n);
  }
  for (int i = 0; i < 8; ++i) {
    int n = G4H * HH;
    k_f32_to_bf16<<<(n + 255) / 256, 256, 0, stream>>>(W_in[i], w_bf[i], n);
  }
  for (int i = 0; i < 4; ++i)
    k_add_vec<<<(G4H + 255) / 256, 256, 0, stream>>>(bih[i], bhh[i],
                                                     bias_all + (size_t)i * G4H, G4H);

  // ---- two stacked bidirectional LSTM layers ----
  for (int layer = 0; layer < 2; ++layer) {
    hipMemsetAsync(Hp,  0, (size_t)2 * BB * HH * 2, stream);
    hipMemsetAsync(Cst, 0, (size_t)2 * BB * HH * 4, stream);

    const unsigned short *Xf, *Xb, *Wihf, *Whhf, *Wihb, *Whhb;
    const float *bf_, *bb_;
    unsigned short *of, *ob;
    if (layer == 0) {
      Xf = x_bf; Xb = x_bf;
      Wihf = w_bf[0]; Whhf = w_bf[1]; Wihb = w_bf[2]; Whhb = w_bf[3];
      bf_ = bias_all; bb_ = bias_all + G4H;
      of = hs0f; ob = hs0b;
    } else {
      Xf = hs0f; Xb = hs0b;
      Wihf = w_bf[4]; Whhf = w_bf[5]; Wihb = w_bf[6]; Whhb = w_bf[7];
      bf_ = bias_all + 2 * G4H; bb_ = bias_all + 3 * G4H;
      of = hs1f; ob = hs1b;
    }

    for (int t = 0; t < SS; ++t) {
      int tf = t, tb = SS - 1 - t;
      k_lstm_gemm<<<128, 256, 0, stream>>>(Xf, Xb, tf, tb,
                                           Wihf, Whhf, Wihb, Whhb,
                                           bf_, bb_, Hp, G);
      k_lstm_cell<<<256, 256, 0, stream>>>(G, Cst, Hp, of, ob, tf, tb);
    }
  }

  // ---- attention pooling ----
  k_attn_scores<<<BB * SS, 64, 0, stream>>>(hs1f, hs1b, Wa1, Wa2, sc);
  k_attn_pool<<<BB, 512, 0, stream>>>(sc, hs1f, hs1b, (float*)d_out);
}
